// BlockedEllLinear_77730318123392
// MI455X (gfx1250) — matrix operand order repaired and checked by reading, not compiled, not run on hardware
//
#include <hip/hip_runtime.h>
#include <stdint.h>

#define TOKENS 8192
#define IN_F   4096
#define OUT_F  4096

#define TM 256          // output tile rows (tokens) per workgroup
#define TN 128          // output tile cols (out features) per workgroup
#define BK 32           // K step per LDS stage (one bf16 WMMA K)
#define KP 40           // padded LDS row stride in elements (bank-conflict free)
#define NSTAGE 3        // async pipeline depth (ring buffer)

typedef __attribute__((ext_vector_type(16))) __bf16 bf16x16;
typedef __attribute__((ext_vector_type(8)))  __bf16 bf16x8;
typedef __attribute__((ext_vector_type(8)))  float  f32x8;
typedef __attribute__((ext_vector_type(4)))  float  f32x4;   // native vector for NT loads

// Pointee type matching the async builtin's parameter: int __vector(4)
typedef int v4i_t __attribute__((vector_size(4 * sizeof(int))));
typedef __attribute__((address_space(1))) v4i_t* g_v4i_ptr;
typedef __attribute__((address_space(3))) v4i_t* l_v4i_ptr;

// ---------------------------------------------------------------------------
// CDNA5 async global->LDS copy (ASYNCcnt-tracked), 16 bytes per lane.
// ---------------------------------------------------------------------------
__device__ __forceinline__ void async_g2l_b128(const void* g, void* l) {
#if __has_builtin(__builtin_amdgcn_global_load_async_to_lds_b128)
    g_v4i_ptr gp = (g_v4i_ptr)(uintptr_t)g;
    l_v4i_ptr lp = (l_v4i_ptr)(uint32_t)(uintptr_t)l;  // low 32 bits = DS offset
    __builtin_amdgcn_global_load_async_to_lds_b128(gp, lp, 0, 0);
#else
    unsigned lds_off = (unsigned)(uintptr_t)l;
    asm volatile("global_load_async_to_lds_b128 %0, %1, off"
                 :: "v"(lds_off), "v"(g) : "memory");
#endif
}

// Wait until at most one stage (6 async ops) remains outstanding.
__device__ __forceinline__ void wait_async_le6() {
#if __has_builtin(__builtin_amdgcn_s_wait_asynccnt)
    __builtin_amdgcn_s_wait_asynccnt(6);
#else
    asm volatile("s_wait_asynccnt 6" ::: "memory");
#endif
}

__device__ __forceinline__ void wait_async_le0() {
#if __has_builtin(__builtin_amdgcn_s_wait_asynccnt)
    __builtin_amdgcn_s_wait_asynccnt(0);
#else
    asm volatile("s_wait_asynccnt 0" ::: "memory");
#endif
}

// ---------------------------------------------------------------------------
// Pre-pass 1: X f32 -> bf16 (8 elements per thread); NT reads (single use)
// ---------------------------------------------------------------------------
__global__ __launch_bounds__(256) void cvt_x_kernel(const float* __restrict__ x,
                                                    __bf16* __restrict__ xb) {
    size_t i = ((size_t)blockIdx.x * 256 + threadIdx.x) * 8;
    const f32x4 f0 = __builtin_nontemporal_load((const f32x4*)(x + i));
    const f32x4 f1 = __builtin_nontemporal_load((const f32x4*)(x + i + 4));
    bf16x8 o;
    o[0] = (__bf16)f0[0]; o[1] = (__bf16)f0[1]; o[2] = (__bf16)f0[2]; o[3] = (__bf16)f0[3];
    o[4] = (__bf16)f1[0]; o[5] = (__bf16)f1[1]; o[6] = (__bf16)f1[2]; o[7] = (__bf16)f1[3];
    *(bf16x8*)(xb + i) = o;   // regular store: bf16 working set should stay in L2
}

// ---------------------------------------------------------------------------
// Pre-pass 2: Wp = W * (1 + mask_block), f32 -> bf16.
// An 8-aligned run of 8 columns never crosses a 16-wide block boundary.
// ---------------------------------------------------------------------------
__global__ __launch_bounds__(256) void cvt_w_kernel(const float* __restrict__ w,
                                                    const int* __restrict__ mask,
                                                    __bf16* __restrict__ wb) {
    size_t i = ((size_t)blockIdx.x * 256 + threadIdx.x) * 8;
    const int row = (int)(i >> 12);          // / IN_F
    const int col = (int)(i & (IN_F - 1));
    const float s = mask[(row >> 4) * (IN_F / 16) + (col >> 4)] ? 2.0f : 1.0f;
    const f32x4 f0 = __builtin_nontemporal_load((const f32x4*)(w + i));
    const f32x4 f1 = __builtin_nontemporal_load((const f32x4*)(w + i + 4));
    bf16x8 o;
    o[0] = (__bf16)(f0[0] * s); o[1] = (__bf16)(f0[1] * s);
    o[2] = (__bf16)(f0[2] * s); o[3] = (__bf16)(f0[3] * s);
    o[4] = (__bf16)(f1[0] * s); o[5] = (__bf16)(f1[1] * s);
    o[6] = (__bf16)(f1[2] * s); o[7] = (__bf16)(f1[3] * s);
    *(bf16x8*)(wb + i) = o;
}

// ---------------------------------------------------------------------------
// Main GEMM: C = Xb @ Wb^T + bias, bf16 inputs, f32 accumulate (WMMA).
// 256 threads = 8 waves in a 4x2 grid; each wave owns a 64x64 sub-tile
// = 4x4 WMMA accumulators. 3-stage async ring buffer keeps 2 stages of
// global->LDS traffic in flight (ASYNCcnt <= 6 waits drain only the oldest).
// ---------------------------------------------------------------------------
__global__ __launch_bounds__(256) void bse_gemm(const __bf16* __restrict__ Xb,
                                                const __bf16* __restrict__ Wb,
                                                const float*  __restrict__ bias,
                                                float* __restrict__ out) {
    __shared__ __align__(16) __bf16 Xs[NSTAGE][TM * KP];   // 3 x 20 KB
    __shared__ __align__(16) __bf16 Ws[NSTAGE][TN * KP];   // 3 x 10 KB

    const int tid  = threadIdx.x;
    const int lane = tid & 31;
    const int wave = tid >> 5;        // 0..7
    const int wm   = wave >> 1;       // 0..3 -> 64-row slab
    const int wn   = wave & 1;        // 0..1 -> 64-col slab
    const int lr   = lane & 15;
    const int lh   = lane >> 4;       // half-wave select

    const int m0 = blockIdx.y * TM;
    const int n0 = blockIdx.x * TN;

    // loader mapping: 256 threads x 6 async b128 = 24 KB per stage (X+W)
    const int lrow = tid >> 2;        // 0..63
    const int lcol = (tid & 3) * 8;   // 0,8,16,24

    f32x8 acc[4][4] = {};

    auto issue = [&](int kt, int buf) {
        const int k0 = kt * BK;
#pragma unroll
        for (int p = 0; p < 4; ++p) {
            const int r = lrow + p * 64;     // 0..255
            async_g2l_b128(Xb + (size_t)(m0 + r) * IN_F + k0 + lcol,
                           &Xs[buf][r * KP + lcol]);
        }
#pragma unroll
        for (int p = 0; p < 2; ++p) {
            const int r = lrow + p * 64;     // 0..127
            async_g2l_b128(Wb + (size_t)(n0 + r) * IN_F + k0 + lcol,
                           &Ws[buf][r * KP + lcol]);
        }
    };

    issue(0, 0);
    issue(1, 1);
    const int NT = IN_F / BK;   // 128 K-stages

    int cur = 0;                // buffer holding stage kt
    for (int kt = 0; kt < NT; ++kt) {
        // Stage kt done; stage kt+1 may still be in flight.
        if (kt + 1 < NT) wait_async_le6();
        else             wait_async_le0();
        __syncthreads();        // all waves' stage-kt data in LDS; buf (kt+2)%3 free
        if (kt + 2 < NT) {
            int pre = cur + 2; if (pre >= NSTAGE) pre -= NSTAGE;
            issue(kt + 2, pre);
        }

        // A fragments: 16x32 bf16, lane lr holds row M=lr,
        // K chunks {lh*8 .. lh*8+7} and {16+lh*8 .. 16+lh*8+7}
        bf16x16 a[4];
#pragma unroll
        for (int mi = 0; mi < 4; ++mi) {
            const __bf16* p = &Xs[cur][(wm * 64 + mi * 16 + lr) * KP + lh * 8];
            bf16x8 lo = *(const bf16x8*)p;
            bf16x8 hi = *(const bf16x8*)(p + 16);
            a[mi] = __builtin_shufflevector(lo, hi,
                     0,1,2,3,4,5,6,7,8,9,10,11,12,13,14,15);
        }
        // B fragments: 32x16 bf16, lane lr holds column N=lr (= W row),
        // K chunks {lh*8 .. lh*8+7} and {16+lh*8 .. 16+lh*8+7}
        bf16x16 b[4];
#pragma unroll
        for (int ni = 0; ni < 4; ++ni) {
            const __bf16* p = &Ws[cur][(wn * 64 + ni * 16 + lr) * KP + lh * 8];
            bf16x8 lo = *(const bf16x8*)p;
            bf16x8 hi = *(const bf16x8*)(p + 16);
            b[ni] = __builtin_shufflevector(lo, hi,
                     0,1,2,3,4,5,6,7,8,9,10,11,12,13,14,15);
        }
#pragma unroll
        for (int mi = 0; mi < 4; ++mi)
#pragma unroll
            for (int ni = 0; ni < 4; ++ni)
                acc[mi][ni] = __builtin_amdgcn_wmma_f32_16x16x32_bf16(
                    false, a[mi], false, b[ni],
                    (short)0, acc[mi][ni], false, false);

        ++cur; if (cur >= NSTAGE) cur = 0;
    }

    // Epilogue: C layout -> VGPR r holds M = r + 8*lh, N = lr (per 16x16 acc).
    // Non-temporal stores: output is write-once, keep bf16 operands in L2.
#pragma unroll
    for (int ni = 0; ni < 4; ++ni) {
        const int o  = n0 + wn * 64 + ni * 16 + lr;
        const float bv = bias[o];
#pragma unroll
        for (int mi = 0; mi < 4; ++mi) {
            const int m = m0 + wm * 64 + mi * 16 + lh * 8;
#pragma unroll
            for (int r = 0; r < 8; ++r)
                __builtin_nontemporal_store(acc[mi][ni][r] + bv,
                                            out + (size_t)(m + r) * OUT_F + o);
        }
    }
}

// ---------------------------------------------------------------------------
extern "C" void kernel_launch(void* const* d_in, const int* in_sizes, int n_in,
                              void* d_out, int out_size, void* d_ws, size_t ws_size,
                              hipStream_t stream) {
    const float* x    = (const float*)d_in[0];
    const float* w    = (const float*)d_in[1];
    const float* bias = (const float*)d_in[2];
    const int*   mask = (const int*)d_in[3];   // bool -> int per harness convention

    __bf16* xb = (__bf16*)d_ws;                                      // 64 MB
    __bf16* wb = (__bf16*)((char*)d_ws + (size_t)TOKENS * IN_F * 2); // 32 MB
    float*  out = (float*)d_out;

    cvt_x_kernel<<<(unsigned)(((size_t)TOKENS * IN_F) / 2048), 256, 0, stream>>>(x, xb);
    cvt_w_kernel<<<(unsigned)(((size_t)OUT_F * IN_F) / 2048), 256, 0, stream>>>(w, mask, wb);
    bse_gemm<<<dim3(OUT_F / TN, TOKENS / TM), 256, 0, stream>>>(xb, wb, bias, out);
}
